// SplitHeadSAMEncoderAttention_1357209666472
// MI455X (gfx1250) — matrix operand order repaired and checked by reading, not compiled, not run on hardware
//
#include <hip/hip_runtime.h>
#include <hip/hip_bf16.h>

typedef __attribute__((ext_vector_type(16))) _Float16 v16h;
typedef __attribute__((ext_vector_type(8)))  _Float16 v8h;
typedef __attribute__((ext_vector_type(8)))  float    v8f;

#define SEQ 4096
#define CH  768
#define NH  12
#define HD  64

// ---- WMMA fragment loaders (contiguous k = 16*hi + s order, wave32) ----

// A operand: 16x32 f16 tile from row-major matrix, row stride ld (elements).
// lane m = lane&15 holds row m; halves h: k = h + 16*(lane>>4)
// => two contiguous 16B chunks per lane at cols 16*hi and 16*hi+8.
__device__ __forceinline__ v16h load_a16(const _Float16* p, int ld, int lane) {
  const int m = lane & 15, hi = lane >> 4;
  const _Float16* base = p + (size_t)m * ld + hi * 16;
  v8h c0 = *(const v8h*)(base);
  v8h c1 = *(const v8h*)(base + 8);
  v16h a;
#pragma unroll
  for (int i = 0; i < 8; ++i) { a[i] = c0[i]; a[i + 8] = c1[i]; }
  return a;
}

// B operand: 32x16 f16 tile where B[k][n] = M[n][k] with M row-major (ld elems).
// lane n = lane&15 holds column n; halves h: k = h + 16*(lane>>4)
// => one contiguous 32B chunk per lane.
__device__ __forceinline__ v16h load_b16(const _Float16* p, int ld, int lane) {
  const int n = lane & 15, hi = lane >> 4;
  return *(const v16h*)(p + (size_t)n * ld + hi * 16);
}

__device__ __forceinline__ v8f wmma_f16(v16h a, v16h b, v8f c) {
  return __builtin_amdgcn_wmma_f32_16x16x32_f16(false, a, false, b,
                                                (short)0, c, false, false);
}

// ---- Kernel 0: fp32 -> f16 convert ----
__global__ void cvt_f32_f16(const float* __restrict__ in,
                            _Float16* __restrict__ out, int nelem) {
  int i = blockIdx.x * blockDim.x + threadIdx.x;
  if (i < nelem) out[i] = (_Float16)in[i];
}

// ---- Kernels 1 & 3: Y(4096x768) = X(4096x768,f16) * W(768x768,(o,c),f16)^T + b
// mode 0: Q  -> y16[(o>>6)][s][o&63], scaled by (1/8)*log2(e)  (exp2 domain)
// mode 1: K  -> y16[(o>>6)][s][o&63]
// mode 2: V  -> y16[o][s]              (transposed: [head][d][s] == [o][s])
// mode 3: out-> y32[s][o]              (fp32 final output)
__global__ __launch_bounds__(128) void gemm_proj(
    const _Float16* __restrict__ X, const _Float16* __restrict__ W,
    const float* __restrict__ bias, _Float16* __restrict__ y16,
    float* __restrict__ y32, const int mode, const float oscale) {
  const int lane = threadIdx.x & 31;
  const int wave = threadIdx.x >> 5;
  const int m0 = blockIdx.x * 64 + wave * 16;
  const int n0 = blockIdx.y * 64;

  v8f acc[4] = {};
  const _Float16* arow = X + (size_t)m0 * CH;

  for (int k0 = 0; k0 < CH; k0 += 32) {
    __builtin_prefetch(arow + k0 + 256, 0, 1);  // global_prefetch_b8
    v16h a = load_a16(arow + k0, CH, lane);
#pragma unroll
    for (int t = 0; t < 4; ++t) {
      v16h b = load_b16(W + (size_t)(n0 + 16 * t) * CH + k0, CH, lane);
      acc[t] = wmma_f16(a, b, acc[t]);
    }
  }

  const int n = lane & 15, hi = lane >> 4;
#pragma unroll
  for (int t = 0; t < 4; ++t) {
    const int o = n0 + 16 * t + n;
    const float bv = bias[o];
#pragma unroll
    for (int r = 0; r < 8; ++r) {
      const int s = m0 + r + 8 * hi;
      const float v = (acc[t][r] + bv) * oscale;
      if (mode == 2) {
        y16[(size_t)o * SEQ + s] = (_Float16)v;
      } else if (mode == 3) {
        y32[(size_t)s * CH + o] = v;
      } else {
        y16[(size_t)(o >> 6) * SEQ * HD + (size_t)s * HD + (o & 63)] =
            (_Float16)v;
      }
    }
  }
}

// ---- Kernel 2: flash attention (exp2-domain online softmax) ----
// Q (NH,SEQ,HD) pre-scaled by scale*log2e; K (NH,SEQ,HD); Vt (NH,HD,SEQ);
// O (SEQ,CH) f16. Block = 4 waves; wave owns a 16-row q strip; kv chunks of 64.
__global__ __launch_bounds__(128) void flash_attn(
    const _Float16* __restrict__ Q, const _Float16* __restrict__ K,
    const _Float16* __restrict__ Vt, _Float16* __restrict__ O) {
  __shared__ __attribute__((aligned(64))) _Float16 pbuf[4][16 * 64];
  const int lane = threadIdx.x & 31;
  const int wave = threadIdx.x >> 5;
  const int head = blockIdx.y;
  const int q0 = blockIdx.x * 64 + wave * 16;
  const int n = lane & 15, hi = lane >> 4;

  const _Float16* Qb = Q + (size_t)head * SEQ * HD;
  const _Float16* Kb = K + (size_t)head * SEQ * HD;
  const _Float16* Vb = Vt + (size_t)head * HD * SEQ;
  _Float16* pw = pbuf[wave];

  // Q strip lives in registers across the whole kv loop (K-dim 64 = 2 chunks)
  const v16h aq0 = load_a16(Qb + (size_t)q0 * HD, HD, lane);
  const v16h aq1 = load_a16(Qb + (size_t)q0 * HD + 32, HD, lane);

  v8f oacc[4] = {};
  float mrow[8], lrow[8];
#pragma unroll
  for (int r = 0; r < 8; ++r) { mrow[r] = -1e30f; lrow[r] = 0.f; }

  for (int kv0 = 0; kv0 < SEQ; kv0 += 64) {
    __builtin_prefetch(Kb + (size_t)(kv0 + 256) * HD, 0, 1);
    __builtin_prefetch(Vb + (size_t)kv0 + 256, 0, 1);

    // scores: four 16x16 tiles covering kv0..kv0+63 (log2-domain: Q prescaled)
    v8f sc[4];
#pragma unroll
    for (int j = 0; j < 4; ++j) {
      v8f z = {};
      z = wmma_f16(aq0, load_b16(Kb + (size_t)(kv0 + 16 * j) * HD, HD, lane), z);
      z = wmma_f16(aq1, load_b16(Kb + (size_t)(kv0 + 16 * j) * HD + 32, HD, lane), z);
      sc[j] = z;
    }

    // online softmax; C-layout: lane holds rows r+8*hi, col n+16j for tile j.
    // shfl_xor masks {1,2,4,8} reduce across the 16-lane half-groups only.
#pragma unroll
    for (int r = 0; r < 8; ++r) {
      float mx = fmaxf(fmaxf(sc[0][r], sc[1][r]), fmaxf(sc[2][r], sc[3][r]));
      mx = fmaxf(mx, __shfl_xor(mx, 1, 32));
      mx = fmaxf(mx, __shfl_xor(mx, 2, 32));
      mx = fmaxf(mx, __shfl_xor(mx, 4, 32));
      mx = fmaxf(mx, __shfl_xor(mx, 8, 32));
      const float mnew = fmaxf(mrow[r], mx);
      const float alpha = __builtin_amdgcn_exp2f(mrow[r] - mnew);
      float p[4];
#pragma unroll
      for (int j = 0; j < 4; ++j)
        p[j] = __builtin_amdgcn_exp2f(sc[j][r] - mnew);
      float rs = (p[0] + p[1]) + (p[2] + p[3]);
      rs += __shfl_xor(rs, 1, 32);
      rs += __shfl_xor(rs, 2, 32);
      rs += __shfl_xor(rs, 4, 32);
      rs += __shfl_xor(rs, 8, 32);
      lrow[r] = lrow[r] * alpha + rs;
      mrow[r] = mnew;
#pragma unroll
      for (int t = 0; t < 4; ++t) oacc[t][r] *= alpha;
      // stage P (row-major 16x64) in LDS to re-swizzle into A-operand layout
#pragma unroll
      for (int j = 0; j < 4; ++j)
        pw[(r + 8 * hi) * 64 + 16 * j + n] = (_Float16)p[j];
    }

    // P(16x64) x V(64x64): B operand from Vt rows (contiguous along kv)
    v16h ap0 = load_a16(pw, 64, lane);
    v16h ap1 = load_a16(pw + 32, 64, lane);
#pragma unroll
    for (int t = 0; t < 4; ++t) {
      oacc[t] = wmma_f16(ap0, load_b16(Vb + (size_t)(16 * t) * SEQ + kv0, SEQ, lane), oacc[t]);
      oacc[t] = wmma_f16(ap1, load_b16(Vb + (size_t)(16 * t) * SEQ + kv0 + 32, SEQ, lane), oacc[t]);
    }
  }

  // normalize and store merged (s, head*64+d) f16 for the output projection
#pragma unroll
  for (int t = 0; t < 4; ++t) {
#pragma unroll
    for (int r = 0; r < 8; ++r) {
      const int s = q0 + r + 8 * hi;
      const float v = oacc[t][r] / lrow[r];
      O[(size_t)s * CH + head * HD + 16 * t + n] = (_Float16)v;
    }
  }
}

extern "C" void kernel_launch(void* const* d_in, const int* in_sizes, int n_in,
                              void* d_out, int out_size, void* d_ws,
                              size_t ws_size, hipStream_t stream) {
  (void)in_sizes; (void)n_in; (void)out_size; (void)ws_size;
  const float* x  = (const float*)d_in[0];
  const float* Wq = (const float*)d_in[1];
  const float* bq = (const float*)d_in[2];
  const float* Wk = (const float*)d_in[3];
  const float* bk = (const float*)d_in[4];
  const float* Wv = (const float*)d_in[5];
  const float* bv = (const float*)d_in[6];
  const float* Wp = (const float*)d_in[7];
  const float* bp = (const float*)d_in[8];

  char* ws = (char*)d_ws;                       // 36.2 MB scratch layout
  _Float16* Xh  = (_Float16*)(ws);              // 4096x768        6291456 B
  _Float16* Wqh = (_Float16*)(ws + 6291456);    // 768x768         1179648 B
  _Float16* Wkh = (_Float16*)(ws + 7471104);
  _Float16* Wvh = (_Float16*)(ws + 8650752);
  _Float16* Wph = (_Float16*)(ws + 9830400);
  _Float16* Qh  = (_Float16*)(ws + 11010048);   // (NH,SEQ,HD)
  _Float16* Kh  = (_Float16*)(ws + 17301504);   // (NH,SEQ,HD)
  _Float16* Vth = (_Float16*)(ws + 23592960);   // (NH,HD,SEQ)
  _Float16* Oh  = (_Float16*)(ws + 29884416);   // (SEQ,CH)

  cvt_f32_f16<<<(SEQ * CH + 255) / 256, 256, 0, stream>>>(x, Xh, SEQ * CH);
  cvt_f32_f16<<<(CH * CH + 255) / 256, 256, 0, stream>>>(Wq, Wqh, CH * CH);
  cvt_f32_f16<<<(CH * CH + 255) / 256, 256, 0, stream>>>(Wk, Wkh, CH * CH);
  cvt_f32_f16<<<(CH * CH + 255) / 256, 256, 0, stream>>>(Wv, Wvh, CH * CH);
  cvt_f32_f16<<<(CH * CH + 255) / 256, 256, 0, stream>>>(Wp, Wph, CH * CH);

  dim3 g(SEQ / 64, CH / 64), b(128);
  // Q epilogue scale = (1/sqrt(64)) * log2(e): exp2-domain flash softmax.
  gemm_proj<<<g, b, 0, stream>>>(Xh, Wqh, bq, Qh, nullptr, 0, 0.18033688f);
  gemm_proj<<<g, b, 0, stream>>>(Xh, Wkh, bk, Kh, nullptr, 1, 1.0f);
  gemm_proj<<<g, b, 0, stream>>>(Xh, Wvh, bv, Vth, nullptr, 2, 1.0f);

  flash_attn<<<dim3(SEQ / 64, NH), 128, 0, stream>>>(Qh, Kh, Vth, Oh);

  gemm_proj<<<g, b, 0, stream>>>(Oh, Wph, bp, nullptr, (float*)d_out, 3, 1.0f);
}
